// HoughVotingDecoder_65790309040388
// MI455X (gfx1250) — matrix operand order repaired and checked by reading, not compiled, not run on hardware
//
#include <hip/hip_runtime.h>
#include <hip/hip_bf16.h>

// CDNA5 wave32 WMMA types
typedef __attribute__((ext_vector_type(16))) _Float16 v16h;
typedef __attribute__((ext_vector_type(8)))  _Float16 v8h;
typedef __attribute__((ext_vector_type(8)))  float    v8f;

#define H_IMG 256
#define W_IMG 320
#define HW_IMG (H_IMG * W_IMG)
#define NCLS 22
#define NB 4
#define VSTEPS 160

// ---------------------------------------------------------------------------
// NCHW f32 -> NHWC f16 transposing convert into the interior of a padded,
// strided concat buffer at channel offset cOff.
// ---------------------------------------------------------------------------
__global__ void k_nchw_to_nhwc_f16(const float* __restrict__ in, _Float16* __restrict__ outInt,
                                   int C, int H, int W, int rowStride, int batchStridePix,
                                   int Ctot, int cOff) {
    int i = blockIdx.x * blockDim.x + threadIdx.x;   // over NB*H*W*C (output index)
    int HW = H * W;
    if (i >= NB * HW * C) return;
    int c = i % C;
    int rest = i / C;
    int p = rest % HW;
    int b = rest / HW;
    int y = p / W, x = p % W;
    outInt[((size_t)b * batchStridePix + y * rowStride + x) * Ctot + cOff + c] =
        (_Float16)in[((size_t)(b * C + c)) * HW + p];
}

// Zero the 1-pixel spatial halo of a padded NHWC buffer [(H+2) x (W+2) x C].
__global__ void k_zero_halo(_Float16* __restrict__ buf, int H, int W, int C) {
    int nh = 2 * (W + 2) + 2 * H;                    // halo pixels per image
    int i = blockIdx.x * blockDim.x + threadIdx.x;   // over NB*nh*C
    if (i >= NB * nh * C) return;
    int c = i % C;
    int rest = i / C;
    int j = rest % nh;
    int b = rest / nh;
    int y, x;
    if (j < W + 2)              { y = 0;      x = j; }
    else if (j < 2 * (W + 2))   { y = H + 1;  x = j - (W + 2); }
    else {
        int k = j - 2 * (W + 2);
        y = 1 + (k >> 1);
        x = (k & 1) ? (W + 1) : 0;
    }
    buf[((size_t)(b * (H + 2) + y) * (W + 2) + x) * C + c] = (_Float16)0.0f;
}

__global__ void k_zero_f32(float* __restrict__ p, int n) {
    int i = blockIdx.x * blockDim.x + threadIdx.x;
    if (i < n) p[i] = 0.0f;
}

// ---------------------------------------------------------------------------
// Repack ConvTranspose2d weights [Cin,Cout,4,4] (torch layout) into a
// per-lane A-matrix layout for V_WMMA_F32_16X16X32_F16 (k = 16*hs + e, same as B), one 16x32
// tile per (quadrant, tap, coTile, ciChunk). Output-parity quadrant (pa,pb)
// of ConvT(k4,s2,p1) uses kernel tap (ky,kx)=(3-2*ty-pa, 3-2*tx-pb) at input
// (iy,ix)=(qa+ty+pa-1, qb+tx+pb-1).
// Packed index = ((((quad*4+tap)*nCoT+coTile)*nChunk+chunk)*32+lane)*16+e
// ---------------------------------------------------------------------------
__global__ void k_pack_convt(const float* __restrict__ w, _Float16* __restrict__ pw,
                             int Cin, int Cout) {
    int idx = blockIdx.x * blockDim.x + threadIdx.x;
    if (idx >= Cin * Cout * 16) return;
    int e    = idx & 15;
    int lane = (idx >> 4) & 31;
    int rest = idx >> 9;
    int nChunk = Cin >> 5;
    int chunk  = rest % nChunk; rest /= nChunk;
    int nCoT   = Cout >> 4;
    int coTile = rest % nCoT;   rest /= nCoT;
    int tap  = rest & 3;
    int quad = rest >> 2;
    int pa = quad >> 1, pb = quad & 1;
    int ty = tap >> 1,  tx = tap & 1;
    int M  = lane & 15;
    int hs = lane >> 4;
    int K  = e + (hs << 4);
    int co = coTile * 16 + M;
    int ci = chunk * 32 + K;
    int kyp = 3 - 2 * ty - pa;
    int kxp = 3 - 2 * tx - pb;
    pw[idx] = (_Float16)w[((ci * Cout + co) * 4 + kyp) * 4 + kxp];
}

// Head weights: rows 0..65 = wv (66x64), 66..87 = wc (22x64), pad to 96 rows.
__global__ void k_pack_head(const float* __restrict__ wv, const float* __restrict__ wc,
                            _Float16* __restrict__ pw) {
    int idx = blockIdx.x * blockDim.x + threadIdx.x;  // 6 coTiles * 2 chunks * 512
    if (idx >= 6144) return;
    int e      = idx & 15;
    int lane   = (idx >> 4) & 31;
    int chunk  = (idx >> 9) & 1;
    int coTile = idx >> 10;
    int M  = lane & 15;
    int hs = lane >> 4;
    int K  = e + (hs << 4);
    int co = coTile * 16 + M;
    int ci = chunk * 32 + K;
    float v = 0.0f;
    if (co < 66)      v = wv[co * 64 + ci];
    else if (co < 88) v = wc[(co - 66) * 64 + ci];
    pw[idx] = (_Float16)v;
}

// ---------------------------------------------------------------------------
// ConvT-as-GEMM on zero-haloed NHWC f16: one wave computes a 16(co) x 16(px)
// f32 tile of one output-parity quadrant. K loop = 4 taps x (Cin/32) WMMA
// chunks; all loads unconditional (halo supplies zeros):
//   per chunk: 2x b128 (packed A) + 2x b128 (NHWC B) + 1 wmma.
// actInt/outInt point at the interior (0,0) pixel of batch 0.
// ---------------------------------------------------------------------------
__global__ void k_convt_wmma(const _Float16* __restrict__ pw,
                             const _Float16* __restrict__ actInt,
                             const float*    __restrict__ bias,
                             _Float16*       __restrict__ outInt,
                             int Cin, int inRow, int inBatch,
                             int Cout, int CoutStride,
                             int Wout, int outRow, int outBatch, int nTiles) {
    int lane = threadIdx.x & 31;
    int wave = threadIdx.x >> 5;
    int tile = blockIdx.x * 4 + wave;      // wave-uniform guard: EXEC all-1 for WMMA
    if (tile >= nTiles) return;
    int coTile = blockIdx.y;
    int bz = blockIdx.z;
    int b = bz >> 2, quad = bz & 3;
    int pa = quad >> 1, pb = quad & 1;
    int hs = lane >> 4, laneN = lane & 15;
    int qW = Wout >> 1;
    int n  = tile * 16 + laneN;            // quadrant pixel index (always in range)
    int qa = n / qW, qb = n % qW;
    int nChunk = Cin >> 5;
    int nCoT   = Cout >> 4;

    v8f acc = {};
    for (int tap = 0; tap < 4; ++tap) {
        int ty = tap >> 1, tx = tap & 1;
        int iy = qa + ty + pa - 1;         // may be -1 / H: lands in zero halo
        int ix = qb + tx + pb - 1;
        const _Float16* pA = pw + (size_t)(((quad * 4 + tap) * nCoT + coTile) * nChunk) * 512
                                + (size_t)lane * 16;
        const _Float16* pB = actInt
            + ((ptrdiff_t)b * inBatch + (ptrdiff_t)iy * inRow + ix) * Cin + (hs << 4);
        for (int chunk = 0; chunk < nChunk; ++chunk) {
            v16h a = *(const v16h*)pA;                 // 2x global_load_b128
            pA += 512;
            v16h bvec = *(const v16h*)(pB + chunk * 32); // 2x global_load_b128
            acc = __builtin_amdgcn_wmma_f32_16x16x32_f16(
                false, a, false, bvec, (short)0, acc, false, false);
        }
    }
    int oy = 2 * qa + pa, ox = 2 * qb + pb;
    v8h res;
#pragma unroll
    for (int r = 0; r < 8; ++r) {
        int co = coTile * 16 + hs * 8 + r;   // D layout: VGPR r -> M=r (+8 for hi lanes)
        float v = acc[r] + bias[co];
        res[r] = (_Float16)(v > 0.0f ? v : 0.0f);
    }
    // NHWC: lane's 8 channels are contiguous -> single b128 store
    *(v8h*)(outInt + ((size_t)b * outBatch + oy * outRow + ox) * CoutStride
                   + coTile * 16 + hs * 8) = res;
}

// ---------------------------------------------------------------------------
// 1x1 head GEMM: 96(padded co) x pixels, K=64 in two WMMA chunks over NHWC y1.
// co<66 -> vertex (f32, NCHW d_out), 66..87 -> logits (f32, ws).
// ---------------------------------------------------------------------------
__global__ void k_head_wmma(const _Float16* __restrict__ pw,
                            const _Float16* __restrict__ act,   // NHWC [B,HW,64]
                            const float* __restrict__ bvv_, const float* __restrict__ bcc_,
                            float* __restrict__ vert, float* __restrict__ logits) {
    int lane = threadIdx.x & 31;
    int wave = threadIdx.x >> 5;
    int tile = blockIdx.x * 4 + wave;
    int coTile = blockIdx.y;
    int b = blockIdx.z;
    int hs = lane >> 4, laneN = lane & 15;
    int n = tile * 16 + laneN;
    const size_t HW = HW_IMG;
    const _Float16* pB = act + ((size_t)b * HW + n) * 64 + (hs << 4);
    v8f acc = {};
#pragma unroll
    for (int chunk = 0; chunk < 2; ++chunk) {
        v16h a = *(const v16h*)(pw + ((size_t)(coTile * 2 + chunk) * 32 + lane) * 16);
        v16h bvec = *(const v16h*)(pB + chunk * 32);
        acc = __builtin_amdgcn_wmma_f32_16x16x32_f16(
            false, a, false, bvec, (short)0, acc, false, false);
    }
#pragma unroll
    for (int r = 0; r < 8; ++r) {
        int co = coTile * 16 + hs * 8 + r;
        float v = acc[r];
        if (co < 66)
            vert[((size_t)b * 66 + co) * HW + n] = v + bvv_[co];
        else if (co < 88)
            logits[((size_t)b * NCLS + (co - 66)) * HW + n] = v + bcc_[co - 66];
    }
}

// argmax over 22 logit channels -> label (as float, first-max tie like jnp)
__global__ void k_label(const float* __restrict__ logits, float* __restrict__ label) {
    int i = blockIdx.x * blockDim.x + threadIdx.x;
    if (i >= NB * HW_IMG) return;
    int b = i / HW_IMG, p = i % HW_IMG;
    float best = logits[(size_t)b * NCLS * HW_IMG + p];
    int bi = 0;
    for (int c = 1; c < NCLS; ++c) {
        float v = logits[((size_t)b * NCLS + c) * HW_IMG + p];
        if (v > best) { best = v; bi = c; }
    }
    label[i] = (float)bi;
}

// ---------------------------------------------------------------------------
// Hough voting: fg pixels (fused seg > 0.5) walk 160 ray steps along their
// normalized direction and atomically accumulate weight. Vote maps
// (4 x 320KB) stay L2-resident; ~50% of pixels skip entirely.
// ---------------------------------------------------------------------------
__global__ void k_vote(const float* __restrict__ seg, const float* __restrict__ vert,
                       float* __restrict__ vmap) {
    int i = blockIdx.x * blockDim.x + threadIdx.x;
    if (i >= NB * HW_IMG) return;
    int b = i / HW_IMG, p = i % HW_IMG;
    float s = 0.0f;
    for (int c = 0; c < NCLS; ++c)
        s += seg[((size_t)b * NCLS + c) * HW_IMG + p];
    if (s <= 0.5f) return;
    float w = fminf(s, 1.0f);
    float dx = vert[((size_t)b * 66 + 0) * HW_IMG + p];
    float dy = vert[((size_t)b * 66 + 1) * HW_IMG + p];
    float inv = 1.0f / (sqrtf(dx * dx + dy * dy) + 1e-6f);
    float u = dx * inv, v = dy * inv;
    float xs = (float)(p % W_IMG), ys = (float)(p / W_IMG);
    float* base = vmap + (size_t)b * HW_IMG;
    for (int t = 1; t <= VSTEPS; ++t) {
        int xt = __float2int_rn(xs + (float)t * u);  // round-nearest-even == jnp.round
        int yt = __float2int_rn(ys + (float)t * v);
        if (xt >= 0 && xt < W_IMG && yt >= 0 && yt < H_IMG)
            atomicAdd(base + yt * W_IMG + xt, w);
    }
}

// per-batch argmax of vote map + masked depth mean -> centers & translations
__global__ void k_center(const float* __restrict__ vmap, const float* __restrict__ seg,
                         const float* __restrict__ depth,
                         const float* __restrict__ fx, const float* __restrict__ fy,
                         const float* __restrict__ px, const float* __restrict__ py,
                         float* __restrict__ centers, float* __restrict__ trans) {
    int b = blockIdx.x;
    __shared__ float sv[256]; __shared__ int si[256];
    __shared__ float sd[256]; __shared__ float sm[256];
    int t = threadIdx.x;
    float best = -1.0f; int bi = 0;
    float dsum = 0.0f, msum = 0.0f;
    for (int p = t; p < HW_IMG; p += 256) {
        float v = vmap[(size_t)b * HW_IMG + p];
        if (v > best) { best = v; bi = p; }
        float s = 0.0f;
        for (int c = 0; c < NCLS; ++c)
            s += seg[((size_t)b * NCLS + c) * HW_IMG + p];
        if (s > 0.5f) { dsum += depth[(size_t)b * HW_IMG + p]; msum += 1.0f; }
    }
    sv[t] = best; si[t] = bi; sd[t] = dsum; sm[t] = msum;
    __syncthreads();
    for (int st = 128; st > 0; st >>= 1) {
        if (t < st) {
            if (sv[t + st] > sv[t] || (sv[t + st] == sv[t] && si[t + st] < si[t])) {
                sv[t] = sv[t + st]; si[t] = si[t + st];
            }
            sd[t] += sd[t + st]; sm[t] += sm[t + st];
        }
        __syncthreads();
    }
    if (t == 0) {
        int ci = si[0];
        float cy = (float)(ci / W_IMG), cx = (float)(ci % W_IMG);
        float tz = sd[0] / (sm[0] + 1e-6f);
        float tx = (cx - px[0]) * tz / fx[0];
        float ty = (cy - py[0]) * tz / fy[0];
        centers[b * 2 + 0] = cx; centers[b * 2 + 1] = cy;
        trans[b * 3 + 0] = tx; trans[b * 3 + 1] = ty; trans[b * 3 + 2] = tz;
    }
}

// ---------------------------------------------------------------------------
extern "C" void kernel_launch(void* const* d_in, const int* in_sizes, int n_in,
                              void* d_out, int out_size, void* d_ws, size_t ws_size,
                              hipStream_t stream) {
    (void)in_sizes; (void)n_in; (void)out_size; (void)ws_size;
    const float* x1    = (const float*)d_in[0];
    const float* x2    = (const float*)d_in[1];
    const float* x3    = (const float*)d_in[2];
    const float* x4    = (const float*)d_in[3];
    const float* x5    = (const float*)d_in[4];
    const float* depth = (const float*)d_in[5];
    const float* seg   = (const float*)d_in[6];
    const float* fx    = (const float*)d_in[7];
    const float* fy    = (const float*)d_in[8];
    const float* px    = (const float*)d_in[9];
    const float* py    = (const float*)d_in[10];
    const float* w5 = (const float*)d_in[11]; const float* b5 = (const float*)d_in[12];
    const float* w4 = (const float*)d_in[13]; const float* b4 = (const float*)d_in[14];
    const float* w3 = (const float*)d_in[15]; const float* b3 = (const float*)d_in[16];
    const float* w2 = (const float*)d_in[17]; const float* b2 = (const float*)d_in[18];
    const float* w1 = (const float*)d_in[19]; const float* b1 = (const float*)d_in[20];
    const float* wv = (const float*)d_in[21]; const float* bv = (const float*)d_in[22];
    const float* wc = (const float*)d_in[23]; const float* bc = (const float*)d_in[24];

    // d_out layout (return order, all f32)
    float* out_vert  = (float*)d_out;                 // [4,66,256,320]
    float* out_label = out_vert + (size_t)NB * 66 * HW_IMG;
    float* out_vmap  = out_label + (size_t)NB * HW_IMG;
    float* out_ctr   = out_vmap + (size_t)NB * HW_IMG;
    float* out_trn   = out_ctr + NB * 2;

    // workspace carve-up (all offsets 256B aligned)
    char* ws = (char*)d_ws;
    size_t off = 0;
    auto alloc = [&](size_t bytes) -> void* {
        void* p = ws + off; off += (bytes + 255) & ~(size_t)255; return p;
    };
    // zero-haloed NHWC f16 concat buffers: [(H+2) x (W+2) x Ctot] per batch
    _Float16* cat1  = (_Float16*)alloc((size_t)NB * 10  * 12  * 512 * 2); // x1
    _Float16* catY5 = (_Float16*)alloc((size_t)NB * 18  * 22  * 512 * 2); // conv5|x2
    _Float16* catY4 = (_Float16*)alloc((size_t)NB * 34  * 42  * 256 * 2); // conv4|x3
    _Float16* catY3 = (_Float16*)alloc((size_t)NB * 66  * 82  * 128 * 2); // conv3|x4
    _Float16* catY2 = (_Float16*)alloc((size_t)NB * 130 * 162 * 128 * 2); // conv2|x5
    _Float16* y1    = (_Float16*)alloc((size_t)NB * 81920 * 64 * 2);      // conv1 (no halo)
    _Float16* pw5 = (_Float16*)alloc((size_t)512 * 256 * 16 * 2);
    _Float16* pw4 = (_Float16*)alloc((size_t)512 * 128 * 16 * 2);
    _Float16* pw3 = (_Float16*)alloc((size_t)256 * 64 * 16 * 2);
    _Float16* pw2 = (_Float16*)alloc((size_t)128 * 64 * 16 * 2);
    _Float16* pw1 = (_Float16*)alloc((size_t)128 * 64 * 16 * 2);
    _Float16* pwh = (_Float16*)alloc((size_t)6144 * 2);
    float* logits = (float*)alloc((size_t)NB * NCLS * HW_IMG * 4);

    // interior-origin pointers (skip one padded row + one column)
    _Float16* cat1I  = cat1  + (size_t)(12 + 1) * 512;
    _Float16* catY5I = catY5 + (size_t)(22 + 1) * 512;
    _Float16* catY4I = catY4 + (size_t)(42 + 1) * 256;
    _Float16* catY3I = catY3 + (size_t)(82 + 1) * 128;
    _Float16* catY2I = catY2 + (size_t)(162 + 1) * 128;

    const int T = 256;
    auto nb = [](int n, int t) { return (n + t - 1) / t; };

    // 0) zero halos (supplies exact zeros for out-of-bounds ConvT taps)
    k_zero_halo<<<nb(NB * 40 * 512, T), T, 0, stream>>>(cat1, 8, 10, 512);
    k_zero_halo<<<nb(NB * 76 * 512, T), T, 0, stream>>>(catY5, 16, 20, 512);
    k_zero_halo<<<nb(NB * 148 * 256, T), T, 0, stream>>>(catY4, 32, 40, 256);
    k_zero_halo<<<nb(NB * 292 * 128, T), T, 0, stream>>>(catY3, 64, 80, 128);
    k_zero_halo<<<nb(NB * 580 * 128, T), T, 0, stream>>>(catY2, 128, 160, 128);

    // 1) inputs/skips -> NHWC f16 interiors of the concat buffers
    k_nchw_to_nhwc_f16<<<nb(NB * 80 * 512, T), T, 0, stream>>>(
        x1, cat1I, 512, 8, 10, 12, 120, 512, 0);
    k_nchw_to_nhwc_f16<<<nb(NB * 320 * 256, T), T, 0, stream>>>(
        x2, catY5I, 256, 16, 20, 22, 396, 512, 256);
    k_nchw_to_nhwc_f16<<<nb(NB * 1280 * 128, T), T, 0, stream>>>(
        x3, catY4I, 128, 32, 40, 42, 1428, 256, 128);
    k_nchw_to_nhwc_f16<<<nb(NB * 5120 * 64, T), T, 0, stream>>>(
        x4, catY3I, 64, 64, 80, 82, 5412, 128, 64);
    k_nchw_to_nhwc_f16<<<nb(NB * 20480 * 64, T), T, 0, stream>>>(
        x5, catY2I, 64, 128, 160, 162, 21060, 128, 64);

    // 2) pack weights into WMMA A-lane layout
    k_pack_convt<<<nb(512 * 256 * 16, T), T, 0, stream>>>(w5, pw5, 512, 256);
    k_pack_convt<<<nb(512 * 128 * 16, T), T, 0, stream>>>(w4, pw4, 512, 128);
    k_pack_convt<<<nb(256 * 64 * 16, T), T, 0, stream>>>(w3, pw3, 256, 64);
    k_pack_convt<<<nb(128 * 64 * 16, T), T, 0, stream>>>(w2, pw2, 128, 64);
    k_pack_convt<<<nb(128 * 64 * 16, T), T, 0, stream>>>(w1, pw1, 128, 64);
    k_pack_head<<<nb(6144, T), T, 0, stream>>>(wv, wc, pwh);

    // 3) decoder: 5 WMMA ConvT layers (grid = pixTiles/4, coTiles, B*quadrant)
    k_convt_wmma<<<dim3(2, 16, 16), 128, 0, stream>>>(pw5, cat1I, b5, catY5I,
        512, 12, 120, 256, 512, 20, 22, 396, 5);
    k_convt_wmma<<<dim3(5, 8, 16), 128, 0, stream>>>(pw4, catY5I, b4, catY4I,
        512, 22, 396, 128, 256, 40, 42, 1428, 20);
    k_convt_wmma<<<dim3(20, 4, 16), 128, 0, stream>>>(pw3, catY4I, b3, catY3I,
        256, 42, 1428, 64, 128, 80, 82, 5412, 80);
    k_convt_wmma<<<dim3(80, 4, 16), 128, 0, stream>>>(pw2, catY3I, b2, catY2I,
        128, 82, 5412, 64, 128, 160, 162, 21060, 320);
    k_convt_wmma<<<dim3(320, 4, 16), 128, 0, stream>>>(pw1, catY2I, b1, y1,
        128, 162, 21060, 64, 64, 320, 320, 81920, 1280);

    // 4) heads (vertex + logits), labels
    k_head_wmma<<<dim3(1280, 6, NB), 128, 0, stream>>>(pwh, y1, bv, bc, out_vert, logits);
    k_label<<<nb(NB * HW_IMG, T), T, 0, stream>>>(logits, out_label);

    // 5) Hough voting + center/translation
    k_zero_f32<<<nb(NB * HW_IMG, T), T, 0, stream>>>(out_vmap, NB * HW_IMG);
    k_vote<<<nb(NB * HW_IMG, T), T, 0, stream>>>(seg, out_vert, out_vmap);
    k_center<<<NB, 256, 0, stream>>>(out_vmap, seg, depth, fx, fy, px, py, out_ctr, out_trn);
}
